// GlmImageVQVAE_52725018526060
// MI455X (gfx1250) — matrix operand order repaired and checked by reading, not compiled, not run on hardware
//
#include <hip/hip_runtime.h>

// ---------------------------------------------------------------------------
// VQ-VAE nearest-codebook-index for MI455X (gfx1250, wave32, WMMA)
//   z   = 1x1conv(hidden)           [N=16384, E=256]   (fp32 math)
//   zn  = l2norm(z) -> bf16         (workspace)
//   cn  = l2norm(embedding) -> bf16 (workspace)
//   idx = argmax_k  zn . cn_k       (bf16 WMMA sweep, async-to-LDS staging,
//                                    4-way split-K + final combine)
// ---------------------------------------------------------------------------

typedef __attribute__((ext_vector_type(16))) __bf16 v16bf;
typedef __attribute__((ext_vector_type(8)))  __bf16 v8bf;
typedef __attribute__((ext_vector_type(8)))  float  v8f;

union V16 { v16bf v; v8bf h[2]; };

#define NPIX   16384   // B*H*W
#define EDIM   256
#define KCODE  16384
#define CDIM   256
#define HW     1024
#define NSPLIT 4
#define CHROWS 32                    // codebook rows staged per iteration
#define NITER  (KCODE / NSPLIT / CHROWS)   // 128 iterations per block
#define LPITCH 264   // bf16 pitch per row (528B = 4-bank rotation, b128-clean)

// CDNA5 async global->LDS copy (ASYNCcnt), bypasses VGPRs.
__device__ __forceinline__ void async_ld_b128(unsigned lds_off, const void* g) {
    asm volatile("global_load_async_to_lds_b128 %0, %1, off"
                 :: "v"(lds_off), "v"((unsigned long long)(uintptr_t)g)
                 : "memory");
}
__device__ __forceinline__ void wait_async0() {
    asm volatile("s_wait_asynccnt 0x0" ::: "memory");
}

// ---------------------------------------------------------------------------
// Kernel 1: z = X (NCHW) * W^T + b, row-l2-normalize, emit bf16 [NPIX][EDIM]
// ---------------------------------------------------------------------------
__global__ __launch_bounds__(256)
void zq_kernel(const float* __restrict__ X, const float* __restrict__ Wc,
               const float* __restrict__ Bc, __bf16* __restrict__ Z) {
    __shared__ float xt[CDIM][32];
    __shared__ float partial[32][8];
    __shared__ float inv[32];

    const int t   = threadIdx.x;
    const int n0  = blockIdx.x * 32;
    const int b   = n0 >> 10;
    const int hw0 = n0 & (HW - 1);

    const float* xsrc = X + (size_t)b * CDIM * HW + hw0;
    const int p  = t & 31;
    const int cb = t >> 5;
    #pragma unroll 4
    for (int i = 0; i < 32; ++i) {
        int c = cb + i * 8;
        xt[c][p] = xsrc[(size_t)c * HW + p];
    }
    __syncthreads();

    const int e = t;
    float acc[32];
    #pragma unroll
    for (int q = 0; q < 32; ++q) acc[q] = 0.0f;

    const float4* w4 = (const float4*)(Wc + (size_t)e * CDIM);
    for (int cq = 0; cq < CDIM / 4; ++cq) {
        float4 w = w4[cq];
        #pragma unroll
        for (int j = 0; j < 4; ++j) {
            const int c = cq * 4 + j;
            const float wj = (j == 0) ? w.x : (j == 1) ? w.y : (j == 2) ? w.z : w.w;
            #pragma unroll
            for (int q = 0; q < 8; ++q) {
                float4 xv = *(const float4*)&xt[c][q * 4];
                acc[q*4+0] += wj * xv.x;
                acc[q*4+1] += wj * xv.y;
                acc[q*4+2] += wj * xv.z;
                acc[q*4+3] += wj * xv.w;
            }
        }
    }
    const float bias = Bc[e];
    #pragma unroll
    for (int q = 0; q < 32; ++q) acc[q] += bias;

    const int lane = t & 31, wv = t >> 5;
    #pragma unroll
    for (int q = 0; q < 32; ++q) {
        float s = acc[q] * acc[q];
        #pragma unroll
        for (int off = 16; off; off >>= 1) s += __shfl_xor(s, off, 32);
        if (lane == 0) partial[q][wv] = s;
    }
    __syncthreads();
    if (t < 32) {
        float s = 0.0f;
        #pragma unroll
        for (int w2 = 0; w2 < 8; ++w2) s += partial[t][w2];
        inv[t] = 1.0f / fmaxf(sqrtf(s), 1e-12f);
    }
    __syncthreads();
    #pragma unroll 4
    for (int q = 0; q < 32; ++q)
        Z[(size_t)(n0 + q) * EDIM + e] = (__bf16)(acc[q] * inv[q]);
}

// ---------------------------------------------------------------------------
// Kernel 2: codebook row l2-normalize -> bf16 [KCODE][EDIM]
// ---------------------------------------------------------------------------
__global__ __launch_bounds__(256)
void cb_kernel(const float* __restrict__ Emb, __bf16* __restrict__ Cb) {
    __shared__ float partial[8];
    __shared__ float invs;
    const int t = threadIdx.x;
    const int k = blockIdx.x;
    const float v = Emb[(size_t)k * EDIM + t];
    float s = v * v;
    #pragma unroll
    for (int off = 16; off; off >>= 1) s += __shfl_xor(s, off, 32);
    if ((t & 31) == 0) partial[t >> 5] = s;
    __syncthreads();
    if (t == 0) {
        float tot = 0.0f;
        #pragma unroll
        for (int i = 0; i < 8; ++i) tot += partial[i];
        invs = 1.0f / fmaxf(sqrtf(tot), 1e-12f);
    }
    __syncthreads();
    Cb[(size_t)k * EDIM + t] = (__bf16)(v * invs);
}

// ---------------------------------------------------------------------------
// Kernel 3: WMMA sweep. Block = 8 waves, each owns a 16-row M tile with all
// 8 A-fragments register-resident. 32 codebook rows per iteration staged with
// global_load_async_to_lds_b128 into double-buffered LDS (one barrier/iter);
// two independent accumulator chains + software-pipelined B-fragment loads
// (ds-load lookahead of one kk step). 4-way codebook split.
// ---------------------------------------------------------------------------
__global__ __launch_bounds__(256)
void vq_kernel(const __bf16* __restrict__ Z, const __bf16* __restrict__ Cb,
               float* __restrict__ candV, int* __restrict__ candI) {
    __shared__ __align__(16) __bf16 ldsB[2][CHROWS * LPITCH];

    const int t     = threadIdx.x;
    const int lane  = t & 31;
    const int wv    = t >> 5;
    const int half  = (lane >> 4) & 1;
    const int r     = lane & 15;
    const int split = blockIdx.y;                 // 0..3, 4096 codes each
    const int kbase = split * (KCODE / NSPLIT);
    const int m0    = blockIdx.x * 128 + wv * 16;
    const int m     = m0 + r;

    // A fragments (ISA 16-bit A 16x32 layout), register-resident for k=256.
    V16 afrag[8];
    const __bf16* zrow = Z + (size_t)m * EDIM;
    #pragma unroll
    for (int kk = 0; kk < 8; ++kk) {
        afrag[kk].h[0] = *(const v8bf*)(zrow + kk * 32 + half * 8);
        afrag[kk].h[1] = *(const v8bf*)(zrow + kk * 32 + 16 + half * 8);
    }

    float bestv[8];
    int   besti[8];
    #pragma unroll
    for (int v = 0; v < 8; ++v) { bestv[v] = -1e30f; besti[v] = 0; }

    // Async staging map: thread -> (row 0..31, four 16B segments of a 512B row)
    const int srow = t >> 3;          // 0..31
    const int sseg = t & 7;           // 0..7
    const unsigned lbase0 = (unsigned)(uintptr_t)&ldsB[0][0];
    const unsigned lbase1 = (unsigned)(uintptr_t)&ldsB[1][0];
    const __bf16* gsrc0 = Cb + (size_t)(kbase + srow) * EDIM + sseg * 8;

    // Prologue: kick off chunk 0.
    #pragma unroll
    for (int j = 0; j < 4; ++j)
        async_ld_b128(lbase0 + (unsigned)(srow * LPITCH + sseg * 8 + j * 64) * 2,
                      gsrc0 + j * 64);

    for (int it = 0; it < NITER; ++it) {
        wait_async0();        // my async loads for chunk `it` have landed
        __syncthreads();      // everyone's landed; everyone done reading other buf

        if (it + 1 < NITER) { // overlap chunk it+1 transfer with compute
            const unsigned lb = ((it + 1) & 1) ? lbase1 : lbase0;
            const __bf16* g  = gsrc0 + (size_t)(it + 1) * CHROWS * EDIM;
            #pragma unroll
            for (int j = 0; j < 4; ++j)
                async_ld_b128(lb + (unsigned)(srow * LPITCH + sseg * 8 + j * 64) * 2,
                              g + j * 64);
        }

        const __bf16* base = ldsB[it & 1];
        const __bf16* bp0  = base + r * LPITCH + half * 8;          // tile0 col
        const __bf16* bp1  = base + (16 + r) * LPITCH + half * 8;   // tile1 col

        v8f acc0 = {};
        v8f acc1 = {};

        // Software-pipelined B fragments: loads run one kk step ahead of the
        // WMMAs so the ds-load latency is off the critical chain.
        V16 c0, c1, n0f, n1f;
        c0.h[0] = *(const v8bf*)(bp0);
        c0.h[1] = *(const v8bf*)(bp0 + 16);
        c1.h[0] = *(const v8bf*)(bp1);
        c1.h[1] = *(const v8bf*)(bp1 + 16);
        #pragma unroll
        for (int kk = 0; kk < 8; ++kk) {
            if (kk + 1 < 8) {
                const __bf16* q0 = bp0 + (kk + 1) * 32;
                const __bf16* q1 = bp1 + (kk + 1) * 32;
                n0f.h[0] = *(const v8bf*)(q0);
                n0f.h[1] = *(const v8bf*)(q0 + 16);
                n1f.h[0] = *(const v8bf*)(q1);
                n1f.h[1] = *(const v8bf*)(q1 + 16);
            }
            acc0 = __builtin_amdgcn_wmma_f32_16x16x32_bf16(
                       false, afrag[kk].v, false, c0.v, (short)0, acc0, false, false);
            acc1 = __builtin_amdgcn_wmma_f32_16x16x32_bf16(
                       false, afrag[kk].v, false, c1.v, (short)0, acc1, false, false);
            c0 = n0f;
            c1 = n1f;
        }

        const int n0n = kbase + it * CHROWS + r;
        #pragma unroll
        for (int v = 0; v < 8; ++v)
            if (acc0[v] > bestv[v]) { bestv[v] = acc0[v]; besti[v] = n0n; }
        #pragma unroll
        for (int v = 0; v < 8; ++v)
            if (acc1[v] > bestv[v]) { bestv[v] = acc1[v]; besti[v] = n0n + 16; }
    }

    // Reduce best across the 16 columns held by each lane half-group.
    #pragma unroll
    for (int v = 0; v < 8; ++v) {
        #pragma unroll
        for (int off = 1; off < 16; off <<= 1) {
            float ov = __shfl_xor(bestv[v], off, 32);
            int   oi = __shfl_xor(besti[v], off, 32);
            if (ov > bestv[v] || (ov == bestv[v] && oi < besti[v])) {
                bestv[v] = ov; besti[v] = oi;
            }
        }
    }
    if (r == 0) {
        #pragma unroll
        for (int v = 0; v < 8; ++v) {
            const int row = m0 + half * 8 + v;
            candV[split * NPIX + row] = bestv[v];
            candI[split * NPIX + row] = besti[v];
        }
    }
}

// ---------------------------------------------------------------------------
// Kernel 4: combine the NSPLIT candidates per row (indices ascend with split,
// strict > keeps the lowest index on ties, matching argmin-first semantics).
// ---------------------------------------------------------------------------
__global__ __launch_bounds__(256)
void combine_kernel(const float* __restrict__ candV, const int* __restrict__ candI,
                    float* __restrict__ Out) {
    const int mrow = blockIdx.x * 256 + threadIdx.x;
    float bv = candV[mrow];
    int   bi = candI[mrow];
    #pragma unroll
    for (int s = 1; s < NSPLIT; ++s) {
        float v = candV[s * NPIX + mrow];
        int   i = candI[s * NPIX + mrow];
        if (v > bv) { bv = v; bi = i; }
    }
    Out[mrow] = (float)bi;
}

// ---------------------------------------------------------------------------
extern "C" void kernel_launch(void* const* d_in, const int* in_sizes, int n_in,
                              void* d_out, int out_size, void* d_ws, size_t ws_size,
                              hipStream_t stream) {
    const float* X   = (const float*)d_in[0];   // hidden_states [16,256,32,32]
    const float* Emb = (const float*)d_in[1];   // embedding     [16384,256]
    const float* Wc  = (const float*)d_in[2];   // conv_w        [256,256]
    const float* Bc  = (const float*)d_in[3];   // conv_b        [256]

    __bf16* Z    = (__bf16*)d_ws;                       // 8 MB
    __bf16* Cb   = Z + (size_t)NPIX * EDIM;             // 8 MB
    float*  cV   = (float*)(Cb + (size_t)KCODE * EDIM); // 256 KB
    int*    cI   = (int*)(cV + (size_t)NSPLIT * NPIX);  // 256 KB
    float*  Out  = (float*)d_out;

    zq_kernel<<<NPIX / 32, 256, 0, stream>>>(X, Wc, Bc, Z);
    cb_kernel<<<KCODE, 256, 0, stream>>>(Emb, Cb);
    vq_kernel<<<dim3(NPIX / 128, NSPLIT), 256, 0, stream>>>(Z, Cb, cV, cI);
    combine_kernel<<<NPIX / 256, 256, 0, stream>>>(cV, cI, Out);
}
